// NSVQ_27058293965120
// MI455X (gfx1250) — compile-verified, matrix operand order checked by reading, not run on hardware
//
#include <hip/hip_runtime.h>
#include <hip/hip_bf16.h>
#include <cmath>

typedef __attribute__((ext_vector_type(16))) __bf16 v16bf;
typedef __attribute__((ext_vector_type(8)))  __bf16 v8bf;
typedef __attribute__((ext_vector_type(8)))  float  v8f;
typedef int v4i_ __attribute__((vector_size(16)));

#define D_DIM      512           // feature dim (compile-time: A panel in registers)
#define LDS_STRIDE 520           // 512 + 8 elem pad -> 4-bank shift per row
#define TILE_ELEMS (16 * LDS_STRIDE)

// ------------------------------------------------ CDNA5 async-copy helpers

__device__ __forceinline__ void async_copy_b128(const __hip_bfloat16* gsrc,
                                                __hip_bfloat16* ldst) {
#if __has_builtin(__builtin_amdgcn_global_load_async_to_lds_b128)
  __builtin_amdgcn_global_load_async_to_lds_b128(
      (__attribute__((address_space(1))) v4i_*)(void*)gsrc,
      (__attribute__((address_space(3))) v4i_*)(void*)ldst,
      0, 0);
#else
  unsigned int loff =
      (unsigned int)(size_t)(__attribute__((address_space(3))) char*)(void*)ldst;
  asm volatile("global_load_async_to_lds_b128 %0, %1, off"
               :: "v"(loff), "v"(gsrc)
               : "memory");
#endif
}

__device__ __forceinline__ void wait_async0() {
#if __has_builtin(__builtin_amdgcn_s_wait_asynccnt)
  __builtin_amdgcn_s_wait_asynccnt(0);
#else
  asm volatile("s_wait_asynccnt 0x0" ::: "memory");
#endif
}

// ---------------------------------------------------------------- prep kernels

__global__ void prep_w_kernel(const float* __restrict__ weights,
                              float* __restrict__ w, int D) {
  int d = blockIdx.x * blockDim.x + threadIdx.x;
  if (d < D) w[d] = fabsf(weights[d]);
}

__global__ void prep_wi_kernel(const float* __restrict__ input,
                               const float* __restrict__ w,
                               __hip_bfloat16* __restrict__ wi,
                               long total, int D) {
  long i = (long)blockIdx.x * blockDim.x + threadIdx.x;
  if (i < total) {
    int d = (int)(i & (long)(D - 1));          // D is a power of two (512)
    wi[i] = __float2bfloat16(input[i] * w[d]);
  }
}

// one block per code row: weighted bf16 codes + cneg[k] = -0.5*||wc_k||^2
__global__ void prep_wc_kernel(const float* __restrict__ cb,
                               const float* __restrict__ w,
                               __hip_bfloat16* __restrict__ wc,
                               float* __restrict__ cneg, int D) {
  int k   = blockIdx.x;
  int tid = threadIdx.x;
  float s = 0.f;
  for (int d = tid; d < D; d += blockDim.x) {
    float x = cb[(long)k * D + d] * w[d];
    __hip_bfloat16 b = __float2bfloat16(x);
    wc[(long)k * D + d] = b;
    float xr = __bfloat162float(b);            // norm of the rounded value
    s += xr * xr;
  }
  for (int m = 16; m >= 1; m >>= 1) s += __shfl_xor(s, m, 32);
  __shared__ float red[8];
  int wave = tid >> 5;
  if ((tid & 31) == 0) red[wave] = s;
  __syncthreads();
  if (tid == 0) {
    float t = 0.f;
    int nw = blockDim.x >> 5;
    for (int i = 0; i < nw; i++) t += red[i];
    cneg[k] = -0.5f * t;
  }
}

// ------------------------------------------- fused WMMA GEMM + argmin (argmax)
// Block: 256 threads = 8 waves. Waves 0-3 scan codes [0,K/2) for row tiles
// 0..3; waves 4-7 scan [K/2,K) for the same rows (K-split for occupancy).
// Block covers 64 rows; per-iteration the block async-copies one 16-code tile
// per K-half into double-buffered LDS.

__global__ void __launch_bounds__(256)
argmin_wmma_kernel(const __hip_bfloat16* __restrict__ wi,
                   const __hip_bfloat16* __restrict__ wc,
                   const float* __restrict__ cneg,
                   int* __restrict__ idx_out,
                   int N, int K) {
  __shared__ __align__(16) __hip_bfloat16 ldsB[2][2][TILE_ELEMS]; // [buf][khalf]
  __shared__ float sbest[2][64];
  __shared__ int   sidxs[2][64];

  const int tid  = threadIdx.x;
  const int lane = tid & 31;
  const int wave = tid >> 5;
  const int l16  = lane & 15;
  const int half = lane >> 4;
  const int rt   = wave & 3;                   // row tile within block
  const int kh   = wave >> 2;                  // K-half this wave scans
  const int m0   = blockIdx.x * 64 + rt * 16;  // this wave's 16-row tile

  const int khsz = K >> 1;                     // codes per K-half
  const int KH   = kh * khsz;                  // this wave's code base
  const int T    = khsz >> 4;                  // 16-code tiles per half

  // cooperative async-copy of BOTH halves' 16x512 bf16 tiles into buffer `buf`
  // 2048 chunks of 16B; 8 per thread
  auto issue_tiles = [&](int trel, int buf) {
#pragma unroll
    for (int j = 0; j < 8; ++j) {
      int c    = tid + 256 * j;                // 0..2047
      int khl  = c >> 10;                      // which K-half's tile
      int cc   = c & 1023;
      int row  = cc >> 6;
      int col8 = (cc & 63) * 8;
      const __hip_bfloat16* gsrc =
          wc + (long)(khl * khsz + trel * 16 + row) * D_DIM + col8;
      async_copy_b128(gsrc, &ldsB[buf][khl][row * LDS_STRIDE + col8]);
    }
  };

  // ---- preload A panel (16 rows x 512 dims, bf16) into registers -----------
  // 16-bit A layout: lane holds row M = lane%16; lanes 0-15 carry K=0..7 and
  // K=16..23 of each 32-chunk, lanes 16-31 carry K=8..15 and K=24..31.
  const __hip_bfloat16* arow = wi + (long)(m0 + l16) * D_DIM;
  const int klo0 = half ? 8 : 0;
  v16bf areg[16];
#pragma unroll
  for (int kk = 0; kk < 16; ++kk) {
    int klo = kk * 32 + klo0;
    v8bf lo = *(const v8bf*)(arow + klo);
    v8bf hi = *(const v8bf*)(arow + klo + 16);
    areg[kk] = __builtin_shufflevector(lo, hi,
        0, 1, 2, 3, 4, 5, 6, 7, 8, 9, 10, 11, 12, 13, 14, 15);
  }

  float best[8];
  int   bidx[8];
#pragma unroll
  for (int v = 0; v < 8; v++) { best[v] = -3.4e38f; bidx[v] = 0; }

  // prologue: tiles 0 of both halves into buffer 0; prefetch first cneg
  issue_tiles(0, 0);
  float cn = cneg[KH + l16];
  wait_async0();
  __syncthreads();

  for (int t = 0; t < T; ++t) {
    const int n0  = KH + t * 16;
    const int cur = t & 1;

    // start fetching the next tiles into the other buffer (async, overlapped)
    if (t + 1 < T) issue_tiles(t + 1, cur ^ 1);
    // prefetch next tile's cneg (consumed next iteration)
    float cn_next = (t + 1 < T) ? cneg[n0 + 16 + l16] : 0.f;

    // ---- consume current tile: score = wi.wc - 0.5*||wc||^2 via C-init ----
    const __hip_bfloat16* bbase = &ldsB[cur][kh][l16 * LDS_STRIDE];

    // two independent WMMA chains; B fetched two k-steps ahead
    v8bf lo0 = *(const v8bf*)(bbase + klo0);
    v8bf hi0 = *(const v8bf*)(bbase + klo0 + 16);
    v8bf lo1 = *(const v8bf*)(bbase + 32 + klo0);
    v8bf hi1 = *(const v8bf*)(bbase + 32 + klo0 + 16);
    v8f acc0 = {cn, cn, cn, cn, cn, cn, cn, cn};
    v8f acc1 = {0.f, 0.f, 0.f, 0.f, 0.f, 0.f, 0.f, 0.f};
#pragma unroll
    for (int kk = 0; kk < 16; kk += 2) {
      v16bf b0 = __builtin_shufflevector(lo0, hi0,
          0, 1, 2, 3, 4, 5, 6, 7, 8, 9, 10, 11, 12, 13, 14, 15);
      v16bf b1 = __builtin_shufflevector(lo1, hi1,
          0, 1, 2, 3, 4, 5, 6, 7, 8, 9, 10, 11, 12, 13, 14, 15);
      if (kk + 2 < 16) {
        int klo = (kk + 2) * 32 + klo0;
        lo0 = *(const v8bf*)(bbase + klo);
        hi0 = *(const v8bf*)(bbase + klo + 16);
        lo1 = *(const v8bf*)(bbase + klo + 32);
        hi1 = *(const v8bf*)(bbase + klo + 48);
      }
      acc0 = __builtin_amdgcn_wmma_f32_16x16x32_bf16(
          false, areg[kk],     false, b0, (short)0, acc0, false, false);
      acc1 = __builtin_amdgcn_wmma_f32_16x16x32_bf16(
          false, areg[kk + 1], false, b1, (short)0, acc1, false, false);
    }

    int code = n0 + l16;                       // one code per lane slot
#pragma unroll
    for (int v = 0; v < 8; v++) {
      float s = acc0[v] + acc1[v];
      bool c = s > best[v];
      best[v] = c ? s    : best[v];
      bidx[v] = c ? code : bidx[v];
    }
    cn = cn_next;

    // next buffer must be fully resident before anyone flips to it
    wait_async0();
    __syncthreads();
  }

  // ---- argmax across the 16 lanes of each half (ties -> lowest index) ------
#pragma unroll
  for (int m = 1; m < 16; m <<= 1) {
#pragma unroll
    for (int v = 0; v < 8; v++) {
      float ov = __shfl_xor(best[v], m, 32);
      int   oi = __shfl_xor(bidx[v], m, 32);
      bool c = (ov > best[v]) || (ov == best[v] && oi < bidx[v]);
      best[v] = c ? ov : best[v];
      bidx[v] = c ? oi : bidx[v];
    }
  }
  if (l16 == 0) {
    int rbase = rt * 16 + half * 8;            // row within block
#pragma unroll
    for (int v = 0; v < 8; v++) {
      sbest[kh][rbase + v] = best[v];
      sidxs[kh][rbase + v] = bidx[v];
    }
  }
  __syncthreads();

  // ---- merge the two K-halves (half-1 indices are larger: ties -> half 0) --
  if (tid < 64) {
    float b0 = sbest[0][tid], b1 = sbest[1][tid];
    int   i0 = sidxs[0][tid], i1 = sidxs[1][tid];
    idx_out[blockIdx.x * 64 + tid] = (b1 > b0) ? i1 : i0;
  }
}

// ---------------------------------------------------------------- epilogue

__global__ void epilogue_kernel(const float* __restrict__ input,
                                const float* __restrict__ cb,
                                const float* __restrict__ w,
                                const float* __restrict__ rnd,
                                const int* __restrict__ idx,
                                float* __restrict__ out, int D) {
  int row = blockIdx.x;
  int tid = threadIdx.x;
  const float* ir = input + (long)row * D;
  const float* rr = rnd   + (long)row * D;
  const float* cr = cb    + (long)idx[row] * D;

  __shared__ float r1[8], r2[8];
  __shared__ float scale;

  float s1 = 0.f, s2 = 0.f;
  for (int d = tid; d < D; d += blockDim.x) {
    float diff = w[d] * (ir[d] - cr[d]);
    s1 += diff * diff;
    float r = rr[d];
    s2 += r * r;
  }
  for (int m = 16; m >= 1; m >>= 1) {
    s1 += __shfl_xor(s1, m, 32);
    s2 += __shfl_xor(s2, m, 32);
  }
  int wave = tid >> 5;
  if ((tid & 31) == 0) { r1[wave] = s1; r2[wave] = s2; }
  __syncthreads();
  if (tid == 0) {
    float t1 = 0.f, t2 = 0.f;
    int nw = blockDim.x >> 5;
    for (int i = 0; i < nw; i++) { t1 += r1[i]; t2 += r2[i]; }
    scale = sqrtf(t1) / sqrtf(t2) + 1e-12f;      // norm_best/norm_rand + EPS
  }
  __syncthreads();
  float sc = scale;
  for (int d = tid; d < D; d += blockDim.x) {
    out[(long)row * D + d] = ir[d] + sc * rr[d] / (w[d] + 1e-12f);
  }
}

// ---------------------------------------------------------------- launch

static inline size_t alignup(size_t x, size_t a) { return (x + a - 1) & ~(a - 1); }

extern "C" void kernel_launch(void* const* d_in, const int* in_sizes, int n_in,
                              void* d_out, int out_size, void* d_ws, size_t ws_size,
                              hipStream_t stream) {
  const float* input     = (const float*)d_in[0];
  const float* codebooks = (const float*)d_in[1];
  const float* weights   = (const float*)d_in[2];
  const float* rnd       = (const float*)d_in[3];
  float* out = (float*)d_out;

  const int  D = in_sizes[2];                // 512
  const long N = (long)in_sizes[0] / D;      // 16384
  const long K = (long)in_sizes[1] / D;      // 4096

  // workspace carve-out
  char* ws = (char*)d_ws;
  size_t off = 0;
  float* w    = (float*)(ws + off); off = alignup(off + (size_t)D * 4, 256);
  float* cneg = (float*)(ws + off); off = alignup(off + (size_t)K * 4, 256);
  int*   idx  = (int*)  (ws + off); off = alignup(off + (size_t)N * 4, 1024);
  __hip_bfloat16* wi = (__hip_bfloat16*)(ws + off);
  off = alignup(off + (size_t)N * D * 2, 1024);
  __hip_bfloat16* wc = (__hip_bfloat16*)(ws + off);

  prep_w_kernel<<<(D + 255) / 256, 256, 0, stream>>>(weights, w, D);

  long total = N * (long)D;
  prep_wi_kernel<<<(unsigned)((total + 255) / 256), 256, 0, stream>>>(
      input, w, wi, total, D);

  prep_wc_kernel<<<(unsigned)K, 256, 0, stream>>>(codebooks, w, wc, cneg, D);

  argmin_wmma_kernel<<<(unsigned)(N / 64), 256, 0, stream>>>(
      wi, wc, cneg, idx, (int)N, (int)K);

  epilogue_kernel<<<(unsigned)N, 256, 0, stream>>>(
      input, codebooks, w, rnd, idx, out, D);
}
